// DifferentialWordSegmentation_1571958030351
// MI455X (gfx1250) — matrix operand order repaired and checked by reading, not compiled, not run on hardware
//
#include <hip/hip_runtime.h>
#include <hip/hip_bf16.h>
#include <float.h>

// ---- CDNA5 wave32 WMMA types ----
typedef __attribute__((ext_vector_type(16))) __bf16 v16bf;
typedef __attribute__((ext_vector_type(8)))  __bf16 v8bf;
typedef __attribute__((ext_vector_type(8)))  float  v8f;

#define BB   32
#define NN   1024
#define HH   512
#define NSEG 1023            // N-1
#define RM1  (BB * NSEG)     // 32736 rows for GEMM1 (divisible by 16)
#define RM2  (BB * NN)       // 32768 rows for word_enc

static __device__ __forceinline__ __bf16 f2bf(float f) {
    unsigned u = __builtin_bit_cast(unsigned, f);
    unsigned r = (u + 0x7FFFu + ((u >> 16) & 1u)) >> 16;
    unsigned short h = (unsigned short)r;
    return __builtin_bit_cast(__bf16, h);
}

static __device__ __forceinline__ v16bf cat8(v8bf lo, v8bf hi) {
    return __builtin_shufflevector(lo, hi, 0,1,2,3,4,5,6,7,8,9,10,11,12,13,14,15);
}

// -------------------------------------------------------------------------
// Pack a KxNw row-major f32 weight into the bf16 WMMA B-matrix tile layout:
// dst[((kt*ntiles+nt)*32 + lane)*16 + e] = W[kt*32 + (e/8)*16 + (lane/16)*8 + e%8][nt*16 + lane%16]
// so each lane's 16-element B fragment is one contiguous 32-byte load.
// -------------------------------------------------------------------------
__global__ void pack_b_kernel(const float* __restrict__ src, __bf16* __restrict__ dst,
                              int K, int Nw) {
    int idx = blockIdx.x * blockDim.x + threadIdx.x;
    int total = K * Nw;
    if (idx >= total) return;
    int e    = idx & 15;
    int lane = (idx >> 4) & 31;
    int tile = idx >> 9;
    int ntiles = Nw >> 4;
    int nt = tile % ntiles;
    int kt = tile / ntiles;
    int k = kt * 32 + (e >> 3) * 16 + (lane >> 4) * 8 + (e & 7);
    int n = nt * 16 + (lane & 15);
    dst[idx] = f2bf(src[(size_t)k * Nw + n]);
}

// -------------------------------------------------------------------------
// L2-normalize each (b,n) row of segment_rep, emit bf16.
// -------------------------------------------------------------------------
__global__ void __launch_bounds__(128) l2norm_kernel(const float* __restrict__ seg,
                                                     __bf16* __restrict__ rn) {
    int row = blockIdx.x;          // B*N rows
    int tid = threadIdx.x;         // 128 threads, 4 floats each
    const float* p = seg + (size_t)row * HH + tid * 4;
    float4 v = *(const float4*)p;
    __shared__ float red[128];
    red[tid] = v.x * v.x + v.y * v.y + v.z * v.z + v.w * v.w;
    __syncthreads();
    for (int off = 64; off > 0; off >>= 1) {
        if (tid < off) red[tid] += red[tid + off];
        __syncthreads();
    }
    float inv = 1.0f / sqrtf(red[0]);
    __bf16* o = rn + (size_t)row * HH + tid * 4;
    o[0] = f2bf(v.x * inv); o[1] = f2bf(v.y * inv);
    o[2] = f2bf(v.z * inv); o[3] = f2bf(v.w * inv);
}

// -------------------------------------------------------------------------
// GEMM1: S[r] = relu(concat(Rn[b,n],Rn[b,n+1]) @ W1 + b1) @ W2
// One wave per 16-row tile. K = 1024 (two halves from adjacent rows).
// -------------------------------------------------------------------------
__global__ void __launch_bounds__(32) gemm1_kernel(const __bf16* __restrict__ rn,
                                                   const __bf16* __restrict__ pW1,
                                                   const float* __restrict__ b1,
                                                   const float* __restrict__ W2,
                                                   float* __restrict__ S) {
    const int lane = threadIdx.x;
    const int m0   = blockIdx.x * 16;      // 2046 tiles, exact
    const int half = lane >> 4;
    const int lcol = lane & 15;

    int rowm = m0 + lcol;                  // A-matrix row for this lane
    int b = rowm / NSEG;
    int n = rowm - b * NSEG;
    const __bf16* base0 = rn + ((size_t)(b * NN + n)) * HH;
    const __bf16* base1 = base0 + HH;      // Rn[b, n+1]

    float sacc[8];
#pragma unroll
    for (int r = 0; r < 8; ++r) sacc[r] = 0.f;

    for (int nt = 0; nt < 32; ++nt) {
        v8f acc = {};
        for (int kt = 0; kt < 32; ++kt) {
            int kb = kt * 32;
            const __bf16* src = (kb < HH) ? (base0 + kb) : (base1 + (kb - HH));
            v8bf c0 = *(const v8bf*)(src + half * 8);
            v8bf c1 = *(const v8bf*)(src + 16 + half * 8);
            v16bf a  = cat8(c0, c1);
            v16bf bm = *(const v16bf*)(pW1 + (((size_t)kt * 32 + nt) * 32 + lane) * 16);
            acc = __builtin_amdgcn_wmma_f32_16x16x32_bf16(false, a, false, bm,
                                                          (short)0, acc, false, false);
        }
        float bias = b1[nt * 16 + lcol];
        float w2v  = W2[nt * 16 + lcol];
#pragma unroll
        for (int r = 0; r < 8; ++r) {
            float h = acc[r] + bias;
            h = h > 0.f ? h : 0.f;
            sacc[r] += h * w2v;            // dot with W2 slice for this col tile
        }
    }
    // reduce the 16 column-lanes of each half-wave
#pragma unroll
    for (int r = 0; r < 8; ++r) {
        float v = sacc[r];
        v += __shfl_xor(v, 1, 32);
        v += __shfl_xor(v, 2, 32);
        v += __shfl_xor(v, 4, 32);
        v += __shfl_xor(v, 8, 32);
        sacc[r] = v;
    }
    if (lcol == 0) {
        int rbase = m0 + half * 8;
#pragma unroll
        for (int r = 0; r < 8; ++r) S[rbase + r] = sacc[r];
    }
}

// -------------------------------------------------------------------------
// Per-batch: D, peak detect, mask, hard tanh, cumsum, segment labels.
// Emits per-word start index & count (segments are contiguous in n).
// -------------------------------------------------------------------------
__global__ void __launch_bounds__(1024) seg_kernel(const float* __restrict__ S,
                                                   const float* __restrict__ mask,
                                                   int* __restrict__ segcnt,
                                                   int* __restrict__ segstart,
                                                   int* __restrict__ numw) {
    int b = blockIdx.x, tid = threadIdx.x;
    __shared__ float sD[NN];
    __shared__ float sc[NN];
    __shared__ float sc2[NN];
    __shared__ float red[NN];
    __shared__ int   cnts[NN];
    __shared__ int   strts[NN];
    __shared__ float sMin, sMax;

    float s = (tid < NSEG) ? S[b * NSEG + tid] : 0.f;

    red[tid] = (tid < NSEG) ? s : FLT_MAX;
    __syncthreads();
    for (int off = 512; off > 0; off >>= 1) {
        if (tid < off) red[tid] = fminf(red[tid], red[tid + off]);
        __syncthreads();
    }
    if (tid == 0) sMin = red[0];
    __syncthreads();
    red[tid] = (tid < NSEG) ? s : -FLT_MAX;
    __syncthreads();
    for (int off = 512; off > 0; off >>= 1) {
        if (tid < off) red[tid] = fmaxf(red[tid], red[tid + off]);
        __syncthreads();
    }
    if (tid == 0) sMax = red[0];
    __syncthreads();

    sD[tid] = (tid < NSEG) ? (1.f - (s - sMin) / (sMax - sMin)) : 0.f;
    __syncthreads();

    float P = 0.f;
    if (tid < NSEG) {
        float d = sD[tid];
        float fo, so = 0.f;
        if (tid == 0)                           fo = fmaxf(d - sD[1], 0.f);
        else if (tid >= NSEG - 2)               fo = fmaxf(d - sD[tid - 2], 0.f); // final overwrite
        else fo = fminf(fmaxf(d - sD[tid - 1], 0.f), fmaxf(d - sD[tid + 1], 0.f));
        if (tid < 2)                            so = fmaxf(d - sD[tid + 2], 0.f);
        else if (tid < NSEG - 2)
            so = fminf(fmaxf(d - sD[tid - 2], 0.f), fmaxf(d - sD[tid + 2], 0.f));
        P = fminf(fmaxf(fmaxf(fo, so) - 0.05f, 0.f), fo);
    }
    P = fmaxf(P + mask[b * NN + tid] - 1.f, 0.f);   // pad + phn_mask
    sc[tid] = tanhf(100000.f * P);                  // forward value of straight-through
    __syncthreads();

    // inclusive scan (Hillis-Steele, ping-pong)
    float* a = sc; float* c = sc2;
    for (int off = 1; off < NN; off <<= 1) {
        float v = a[tid];
        if (tid >= off) v += a[tid - off];
        c[tid] = v;
        __syncthreads();
        float* t = a; a = c; c = t;
    }

    float addv = (a[0] == 0.f) ? 1.f : 0.f;
    float cv   = a[tid] + addv;
    int   num  = (int)(a[NN - 1] + addv) + 1;

    float rr = roundf(cv);
    int   k  = (int)rr;
    int memberk = (fabsf(cv - rr) < 1e-4f && k >= 1 && k <= num) ? k : 0;

    cnts[tid] = 0; strts[tid] = 1 << 30;
    __syncthreads();
    if (memberk) {
        atomicAdd(&cnts[memberk - 1], 1);
        atomicMin(&strts[memberk - 1], tid);
    }
    __syncthreads();
    segcnt[b * NN + tid]   = cnts[tid];
    segstart[b * NN + tid] = strts[tid];
    if (tid == 0) numw[b] = num;
}

// -------------------------------------------------------------------------
// Segment-mean pooling: word_rep[b,m,:] = mean of member rows (contiguous).
// -------------------------------------------------------------------------
__global__ void __launch_bounds__(128) pool_kernel(const float* __restrict__ seg,
                                                   const int* __restrict__ segcnt,
                                                   const int* __restrict__ segstart,
                                                   __bf16* __restrict__ wrep) {
    int m = blockIdx.x, b = blockIdx.y, tid = threadIdx.x;
    int cnt = segcnt[b * NN + m];
    __bf16* o = wrep + ((size_t)(b * NN + m)) * HH + tid * 4;
    if (cnt == 0) {
        o[0] = o[1] = o[2] = o[3] = f2bf(0.f);
        return;
    }
    int s0 = segstart[b * NN + m];
    float a0 = 0.f, a1 = 0.f, a2 = 0.f, a3 = 0.f;
    const float* p = seg + ((size_t)(b * NN + s0)) * HH + tid * 4;
    for (int j = 0; j < cnt; ++j) {
        float4 v = *(const float4*)p; p += HH;
        a0 += v.x; a1 += v.y; a2 += v.z; a3 += v.w;
    }
    float inv = 1.f / (float)cnt;
    o[0] = f2bf(a0 * inv); o[1] = f2bf(a1 * inv);
    o[2] = f2bf(a2 * inv); o[3] = f2bf(a3 * inv);
}

// -------------------------------------------------------------------------
// word_enc: out = (relu(wrep@We1+be1) @ We2 + be2) * (wrep != 0)
// One wave per 16-row tile; mid activation staged in LDS (16x512 bf16).
// -------------------------------------------------------------------------
__global__ void __launch_bounds__(32) gemm2_kernel(const __bf16* __restrict__ wrep,
                                                   const __bf16* __restrict__ pWe1,
                                                   const __bf16* __restrict__ pWe2,
                                                   const float* __restrict__ be1,
                                                   const float* __restrict__ be2,
                                                   float* __restrict__ out) {
    __shared__ __bf16 hm[16][520];   // padded row stride, 16B-aligned chunks
    const int lane = threadIdx.x;
    const int m0   = blockIdx.x * 16;   // 2048 tiles, exact
    const int half = lane >> 4;
    const int lcol = lane & 15;
    const __bf16* arow = wrep + ((size_t)(m0 + lcol)) * HH;

    for (int nt = 0; nt < 32; ++nt) {
        v8f acc = {};
        for (int kt = 0; kt < 16; ++kt) {
            int kb = kt * 32;
            v8bf c0 = *(const v8bf*)(arow + kb + half * 8);
            v8bf c1 = *(const v8bf*)(arow + kb + 16 + half * 8);
            v16bf a  = cat8(c0, c1);
            v16bf bm = *(const v16bf*)(pWe1 + (((size_t)kt * 32 + nt) * 32 + lane) * 16);
            acc = __builtin_amdgcn_wmma_f32_16x16x32_bf16(false, a, false, bm,
                                                          (short)0, acc, false, false);
        }
        float bias = be1[nt * 16 + lcol];
#pragma unroll
        for (int r = 0; r < 8; ++r) {
            float h = acc[r] + bias;
            h = h > 0.f ? h : 0.f;
            hm[r + half * 8][nt * 16 + lcol] = f2bf(h);
        }
    }
    __syncthreads();

    for (int nt = 0; nt < 32; ++nt) {
        v8f acc = {};
        for (int kt = 0; kt < 16; ++kt) {
            int kb = kt * 32;
            const __bf16* lp = &hm[lcol][kb + half * 8];
            v8bf c0 = *(const v8bf*)(lp);
            v8bf c1 = *(const v8bf*)(lp + 16);
            v16bf a  = cat8(c0, c1);
            v16bf bm = *(const v16bf*)(pWe2 + (((size_t)kt * 32 + nt) * 32 + lane) * 16);
            acc = __builtin_amdgcn_wmma_f32_16x16x32_bf16(false, a, false, bm,
                                                          (short)0, acc, false, false);
        }
        float bias = be2[nt * 16 + lcol];
#pragma unroll
        for (int r = 0; r < 8; ++r) {
            int row = m0 + r + half * 8;
            int col = nt * 16 + lcol;
            unsigned short wm =
                __builtin_bit_cast(unsigned short, wrep[(size_t)row * HH + col]);
            float v = ((wm & 0x7FFF) == 0) ? 0.f : (acc[r] + bias);
            out[(size_t)row * HH + col] = v;
        }
    }
}

// -------------------------------------------------------------------------
extern "C" void kernel_launch(void* const* d_in, const int* in_sizes, int n_in,
                              void* d_out, int out_size, void* d_ws, size_t ws_size,
                              hipStream_t stream) {
    const float* seg  = (const float*)d_in[0];   // (B,N,H)
    const float* mask = (const float*)d_in[1];   // (B,N)
    const float* W1   = (const float*)d_in[2];   // (2H,H)
    const float* b1   = (const float*)d_in[3];   // (H)
    const float* W2   = (const float*)d_in[4];   // (H,1)
    const float* We1  = (const float*)d_in[5];   // (H,H)
    const float* be1  = (const float*)d_in[6];   // (H)
    const float* We2  = (const float*)d_in[7];   // (H,H)
    const float* be2  = (const float*)d_in[8];   // (H)
    float* out = (float*)d_out;                  // (B,N,H)

    // workspace carve-up (256B aligned regions)
    char* ws = (char*)d_ws;
    size_t off = 0;
    auto carve = [&](size_t bytes) -> void* {
        void* p = ws + off;
        off += (bytes + 255) & ~(size_t)255;
        return p;
    };
    __bf16* rn    = (__bf16*)carve((size_t)BB * NN * HH * 2);   // 32MB normalized reps
    __bf16* pW1   = (__bf16*)carve((size_t)2 * HH * HH * 2);    // 1MB
    __bf16* pWe1  = (__bf16*)carve((size_t)HH * HH * 2);        // 512KB
    __bf16* pWe2  = (__bf16*)carve((size_t)HH * HH * 2);        // 512KB
    float*  Sbuf  = (float*)carve((size_t)RM1 * 4);             // similarity scores
    int*    segc  = (int*)carve((size_t)BB * NN * 4);
    int*    segs  = (int*)carve((size_t)BB * NN * 4);
    int*    numw  = (int*)carve((size_t)BB * 4);
    __bf16* wrep  = (__bf16*)carve((size_t)BB * NN * HH * 2);   // 32MB word reps
    (void)ws_size; (void)in_sizes; (void)n_in; (void)out_size;

    // 1) pack weights into WMMA B-matrix layout (bf16)
    pack_b_kernel<<<(2 * HH * HH + 255) / 256, 256, 0, stream>>>(W1, pW1, 2 * HH, HH);
    pack_b_kernel<<<(HH * HH + 255) / 256, 256, 0, stream>>>(We1, pWe1, HH, HH);
    pack_b_kernel<<<(HH * HH + 255) / 256, 256, 0, stream>>>(We2, pWe2, HH, HH);

    // 2) l2-normalize segment reps
    l2norm_kernel<<<BB * NN, 128, 0, stream>>>(seg, rn);

    // 3) similarity MLP (WMMA)
    gemm1_kernel<<<RM1 / 16, 32, 0, stream>>>(rn, pW1, b1, W2, Sbuf);

    // 4) peak detection + scan + segment labeling
    seg_kernel<<<BB, NN, 0, stream>>>(Sbuf, mask, segc, segs, numw);

    // 5) segment-mean pooling -> word_rep (bf16)
    pool_kernel<<<dim3(NN, BB), 128, 0, stream>>>(seg, segc, segs, wrep);

    // 6) word encoder MLP (WMMA x2, LDS-staged) + zero mask
    gemm2_kernel<<<RM2 / 16, 32, 0, stream>>>(wrep, pWe1, pWe2, be1, be2, out);
}
